// Mamba_Block_24532853194769
// MI455X (gfx1250) — matrix operand order repaired and checked by reading, not compiled, not run on hardware
//
#include <hip/hip_runtime.h>
#include <hip/hip_bf16.h>
#include <math.h>

// ---------------------------------------------------------------------------
// Mamba SS2D block for MI455X (gfx1250, wave32, WMMA).
// GEMMs use v_wmma_f32_16x16x32_f16; scan uses wave-per-channel with lane=state.
// ---------------------------------------------------------------------------

typedef __attribute__((ext_vector_type(16))) _Float16 v16h;
typedef __attribute__((ext_vector_type(8)))  _Float16 v8h;
typedef __attribute__((ext_vector_type(8)))  float    v8f;
typedef __attribute__((ext_vector_type(4)))  float    v4f;

#define B_    2
#define H_    48
#define W_    48
#define C_    96
#define D_    192
#define NST   32
#define RK    6
#define K_    4
#define L_    (H_ * W_)          // 2304
#define BL_   (B_ * L_)          // 4608
#define E2_   (2 * D_)           // 384
#define CDBLP 80                 // 70 (dt_rank + 2*d_state) padded to 80

// ---------------- WMMA fragment helpers (ISA 7.12.2 layouts) ----------------
// A 16x32 f16: lane t: m = t&15, khalf = t>>4; elems 0..7 = K khalf*8+0..7,
// elems 8..15 = K 16+khalf*8+0..7.
__device__ __forceinline__ v16h afrag_f16(const _Float16* row, int k0, int kh) {
    v8h lo = *(const v8h*)(row + k0 + kh * 8);
    v8h hi = *(const v8h*)(row + k0 + 16 + kh * 8);
    v16h r;
#pragma unroll
    for (int i = 0; i < 8; ++i) { r[i] = lo[i]; r[i + 8] = hi[i]; }
    return r;
}
__device__ __forceinline__ v16h afrag_f32(const float* row, int k0, int kh) {
    v4f a0 = *(const v4f*)(row + k0 + kh * 8);
    v4f a1 = *(const v4f*)(row + k0 + kh * 8 + 4);
    v4f b0 = *(const v4f*)(row + k0 + 16 + kh * 8);
    v4f b1 = *(const v4f*)(row + k0 + 16 + kh * 8 + 4);
    v16h r;
#pragma unroll
    for (int i = 0; i < 4; ++i) {
        r[i]      = (_Float16)a0[i];
        r[i + 4]  = (_Float16)a1[i];
        r[i + 8]  = (_Float16)b0[i];
        r[i + 12] = (_Float16)b1[i];
    }
    return r;
}
// B 32x16 f16: lane t: n = t&15; bfrag_f32 uses A's K order; bfrag_f16 holds 16 contiguous K at (t>>4)*16.
__device__ __forceinline__ v16h bfrag_f16(const _Float16* row, int k0, int kh) {
    v8h lo = *(const v8h*)(row + k0 + kh * 16);
    v8h hi = *(const v8h*)(row + k0 + kh * 16 + 8);
    v16h r;
#pragma unroll
    for (int i = 0; i < 8; ++i) { r[i] = lo[i]; r[i + 8] = hi[i]; }
    return r;
}
__device__ __forceinline__ v16h bfrag_f32(const float* row, int k0, int kh) {
    const float* p = row + k0 + kh * 8;
    v4f a0 = *(const v4f*)(p);
    v4f a1 = *(const v4f*)(p + 4);
    v4f a2 = *(const v4f*)(p + 16);
    v4f a3 = *(const v4f*)(p + 20);
    v16h r;
#pragma unroll
    for (int i = 0; i < 4; ++i) {
        r[i]      = (_Float16)a0[i];
        r[i + 4]  = (_Float16)a1[i];
        r[i + 8]  = (_Float16)a2[i];
        r[i + 12] = (_Float16)a3[i];
    }
    return r;
}

__device__ __forceinline__ v8f wmma16(v16h a, v16h b, v8f c) {
    // 8 args: (neg_a, A, neg_b, B, c_mod, C, reuse_a, reuse_b)
    return __builtin_amdgcn_wmma_f32_16x16x32_f16(false, a, false, b,
                                                  (short)0, c, false, false);
}

// scan direction k, scan position l -> row-major pixel index p
__device__ __forceinline__ int pmap(int k, int l) {
    if (k == 0) return l;
    if (k == 1) { int w = l / H_, h = l % H_; return h * W_ + w; }
    if (k == 2) return L_ - 1 - l;
    int lp = L_ - 1 - l;
    int w = lp / H_, h = lp % H_;
    return h * W_ + w;
}

__device__ __forceinline__ float silu_f(float x) {
    return x / (1.0f + __expf(-x));
}

// ---------------- Kernel 1: in_proj GEMM (M=4608,N=384,K=96) ----------------
__global__ __launch_bounds__(128) void k_inproj(const float* __restrict__ x,
                                                const float* __restrict__ w,
                                                float* __restrict__ xz) {
    int wave = blockIdx.x * 4 + threadIdx.y;     // 6912 waves
    int nt = wave % (E2_ / 16);                  // 24 N-tiles
    int mt = wave / (E2_ / 16);                  // 288 M-tiles
    int lane = threadIdx.x, q = lane & 15, kh = lane >> 4;
    const float* arow = x + (mt * 16 + q) * C_;
    const float* brow = w + (nt * 16 + q) * C_;
    v8f acc = {};
#pragma unroll
    for (int k0 = 0; k0 < C_; k0 += 32)
        acc = wmma16(afrag_f32(arow, k0, kh), bfrag_f32(brow, k0, kh), acc);
    int m0 = mt * 16 + 8 * kh, n = nt * 16 + q;
#pragma unroll
    for (int r = 0; r < 8; ++r) xz[(m0 + r) * E2_ + n] = acc[r];
}

// ---------------- Kernel 2: depthwise conv 3x3 + SiLU -----------------------
__global__ __launch_bounds__(256) void k_conv(const float* __restrict__ xz,
                                              const float* __restrict__ cw,
                                              const float* __restrict__ cb,
                                              float* __restrict__ cf32,
                                              _Float16* __restrict__ cf16) {
    int idx = blockIdx.x * 256 + threadIdx.x;    // BL_*D_ exactly
    int d = idx % D_;
    int t = idx / D_;
    int p = t % L_, b = t / L_;
    int h = p / W_, w = p % W_;
    float acc = cb[d];
#pragma unroll
    for (int dh = -1; dh <= 1; ++dh)
#pragma unroll
        for (int dw = -1; dw <= 1; ++dw) {
            int hh = h + dh, ww = w + dw;
            if (hh >= 0 && hh < H_ && ww >= 0 && ww < W_)
                acc += cw[d * 9 + (dh + 1) * 3 + (dw + 1)] *
                       xz[(b * L_ + hh * W_ + ww) * E2_ + d];
        }
    float s = silu_f(acc);
    cf32[idx] = s;
    cf16[idx] = (_Float16)s;
}

// ---- Kernel 3: x_dbl GEMM per (b,k): M=2304, N=80(pad of 70), K=192 --------
__global__ __launch_bounds__(128) void k_xdbl(const _Float16* __restrict__ cf16,
                                              const float* __restrict__ xpw,
                                              float* __restrict__ xdbl) {
    int wave = blockIdx.x * 4 + threadIdx.y;     // 5760 waves
    int nt = wave % 5;
    int t  = wave / 5;
    int mt = t % (L_ / 16);                      // 144
    t /= (L_ / 16);
    int k = t % K_, b = t / K_;
    int lane = threadIdx.x, q = lane & 15, kh = lane >> 4;
    int l = mt * 16 + q;
    const _Float16* arow = cf16 + ((size_t)(b * L_ + pmap(k, l))) * D_;
    int c = nt * 16 + q; if (c > 69) c = 69;     // clamp pad rows in-bounds
    const float* brow = xpw + (k * 70 + c) * D_;
    v8f acc = {};
#pragma unroll
    for (int k0 = 0; k0 < D_; k0 += 32)
        acc = wmma16(afrag_f16(arow, k0, kh), bfrag_f32(brow, k0, kh), acc);
    float* out = xdbl + ((size_t)(b * K_ + k) * L_) * CDBLP;
    int l0 = mt * 16 + 8 * kh, n = nt * 16 + q;
#pragma unroll
    for (int r = 0; r < 8; ++r) out[(l0 + r) * CDBLP + n] = acc[r];
}

// ---- Kernel 4: selective scan; wave per (b,k,d), lane per state n ----------
__global__ __launch_bounds__(256) void k_scan(const float* __restrict__ xdbl,
                                              const float* __restrict__ cf32,
                                              const float* __restrict__ dtw,
                                              const float* __restrict__ dtb,
                                              const float* __restrict__ alog,
                                              const float* __restrict__ Ds,
                                              float* __restrict__ ys) {
    int blk = blockIdx.x;                        // 192 blocks
    int dblk = blk % (D_ / 8);
    int t = blk / (D_ / 8);
    int k = t % K_, b = t / K_;
    int d = dblk * 8 + threadIdx.y;
    int lane = threadIdx.x;

    float An = -__expf(alog[(k * D_ + d) * NST + lane]);
    float Dd = Ds[k * D_ + d];
    float bias = dtb[k * D_ + d];
    float wdt[RK];
#pragma unroll
    for (int r = 0; r < RK; ++r) wdt[r] = dtw[(k * D_ + d) * RK + r];

    const float* xrow = xdbl + ((size_t)(b * K_ + k) * L_) * CDBLP;
    float* yout = ys + ((size_t)((b * K_ + k) * D_ + d)) * L_;

    float h = 0.0f;
    for (int l = 0; l < L_; ++l) {
        const float* row = xrow + l * CDBLP;
        __builtin_prefetch(row + 16 * CDBLP, 0, 3);  // global_prefetch_b8
        // fused dt_rank projection + softplus (uniform across lanes)
        float xdt = bias;
#pragma unroll
        for (int r = 0; r < RK; ++r) xdt += row[r] * wdt[r];
        float delta = (xdt > 20.0f) ? xdt : log1pf(__expf(xdt));
        float u = cf32[(b * L_ + pmap(k, l)) * D_ + d];
        float Bn = row[RK + lane];            // c = 6 + n
        float Cn = row[RK + NST + lane];      // c = 38 + n
        h = h * __expf(delta * An) + (delta * u) * Bn;
        float r2 = h * Cn;                    // y = sum_n h*C
        r2 += __shfl_xor(r2, 16, 32);
        r2 += __shfl_xor(r2, 8, 32);
        r2 += __shfl_xor(r2, 4, 32);
        r2 += __shfl_xor(r2, 2, 32);
        r2 += __shfl_xor(r2, 1, 32);
        if (lane == 0) yout[l] = r2 + Dd * u;
    }
}

// ---- Kernel 5: merge 4 directions + LayerNorm + SiLU(z) gate ---------------
__global__ __launch_bounds__(256) void k_combine(const float* __restrict__ ys,
                                                 const float* __restrict__ xz,
                                                 const float* __restrict__ lnw,
                                                 const float* __restrict__ lnb,
                                                 _Float16* __restrict__ yg) {
    int row = blockIdx.x * 8 + threadIdx.y;      // 4608 rows (b,p)
    int b = row / L_, p = row % L_;
    int h = p / W_, w = p % W_;
    int l1 = w * H_ + h;
    int lane = threadIdx.x;

    float vals[6], s = 0.0f, ss = 0.0f;
#pragma unroll
    for (int j = 0; j < 6; ++j) {
        int d = lane + 32 * j;
        float v = ys[((size_t)((b * K_ + 0) * D_ + d)) * L_ + p]
                + ys[((size_t)((b * K_ + 2) * D_ + d)) * L_ + (L_ - 1 - p)]
                + ys[((size_t)((b * K_ + 1) * D_ + d)) * L_ + l1]
                + ys[((size_t)((b * K_ + 3) * D_ + d)) * L_ + (L_ - 1 - l1)];
        vals[j] = v; s += v; ss += v * v;
    }
#pragma unroll
    for (int off = 16; off >= 1; off >>= 1) {
        s  += __shfl_xor(s,  off, 32);
        ss += __shfl_xor(ss, off, 32);
    }
    float mean = s * (1.0f / (float)D_);
    float var  = ss * (1.0f / (float)D_) - mean * mean;
    float rstd = rsqrtf(var + 1e-5f);
#pragma unroll
    for (int j = 0; j < 6; ++j) {
        int d = lane + 32 * j;
        float z = xz[(b * L_ + p) * E2_ + D_ + d];
        float o = (vals[j] - mean) * rstd * lnw[d] + lnb[d];
        yg[(b * L_ + p) * D_ + d] = (_Float16)(o * silu_f(z));
    }
}

// ---- Kernel 6: out_proj GEMM (M=4608,N=96,K=192) + residual ----------------
__global__ __launch_bounds__(128) void k_outproj(const _Float16* __restrict__ yg,
                                                 const float* __restrict__ w,
                                                 const float* __restrict__ xin,
                                                 float* __restrict__ out) {
    int wave = blockIdx.x * 4 + threadIdx.y;     // 1728 waves
    int nt = wave % (C_ / 16);                   // 6
    int mt = wave / (C_ / 16);                   // 288
    int lane = threadIdx.x, q = lane & 15, kh = lane >> 4;
    const _Float16* arow = yg + (mt * 16 + q) * D_;
    const float*    brow = w  + (nt * 16 + q) * D_;
    v8f acc = {};
#pragma unroll
    for (int k0 = 0; k0 < D_; k0 += 32)
        acc = wmma16(afrag_f16(arow, k0, kh), bfrag_f32(brow, k0, kh), acc);
    int m0 = mt * 16 + 8 * kh, n = nt * 16 + q;
#pragma unroll
    for (int r = 0; r < 8; ++r) {
        int m = m0 + r;
        out[m * C_ + n] = xin[m * C_ + n] + acc[r];
    }
}

// ---------------------------- launch ----------------------------------------
extern "C" void kernel_launch(void* const* d_in, const int* in_sizes, int n_in,
                              void* d_out, int out_size, void* d_ws, size_t ws_size,
                              hipStream_t stream) {
    (void)in_sizes; (void)n_in; (void)out_size; (void)ws_size;
    const float* x    = (const float*)d_in[0];
    const float* ipw  = (const float*)d_in[1];
    const float* cw   = (const float*)d_in[2];
    const float* cb   = (const float*)d_in[3];
    const float* xpw  = (const float*)d_in[4];
    const float* dtw  = (const float*)d_in[5];
    const float* dtb  = (const float*)d_in[6];
    const float* alog = (const float*)d_in[7];
    const float* Ds   = (const float*)d_in[8];
    const float* lnw  = (const float*)d_in[9];
    const float* lnb  = (const float*)d_in[10];
    const float* opw  = (const float*)d_in[11];
    float* out = (float*)d_out;

    // workspace layout (all 256B-aligned offsets)
    constexpr size_t XZ_B   = (size_t)BL_ * E2_ * 4;          // 7,077,888
    constexpr size_t CF32_B = (size_t)BL_ * D_ * 4;           // 3,538,944
    constexpr size_t CF16_B = (size_t)BL_ * D_ * 2;           // 1,769,472
    constexpr size_t XDBL_B = (size_t)B_ * K_ * L_ * CDBLP * 4; // 5,898,240
    char* ws = (char*)d_ws;
    float*    xz   = (float*)(ws);
    float*    cf32 = (float*)(ws + XZ_B);
    _Float16* cf16 = (_Float16*)(ws + XZ_B + CF32_B);
    float*    xdbl = (float*)(ws + XZ_B + CF32_B + CF16_B);
    float*    ys   = (float*)(ws + XZ_B + CF32_B + CF16_B + XDBL_B);
    _Float16* yg   = (_Float16*)(ws + XZ_B + CF32_B + CF16_B + XDBL_B +
                                 (size_t)B_ * K_ * D_ * L_ * 4);

    k_inproj <<<(288 * 24) / 4, dim3(32, 4), 0, stream>>>(x, ipw, xz);
    k_conv   <<<(BL_ * D_) / 256, 256, 0, stream>>>(xz, cw, cb, cf32, cf16);
    k_xdbl   <<<(B_ * K_ * 144 * 5) / 4, dim3(32, 4), 0, stream>>>(cf16, xpw, xdbl);
    k_scan   <<<B_ * K_ * (D_ / 8), dim3(32, 8), 0, stream>>>(xdbl, cf32, dtw, dtb,
                                                              alog, Ds, ys);
    k_combine<<<BL_ / 8, dim3(32, 8), 0, stream>>>(ys, xz, lnw, lnb, yg);
    k_outproj<<<(288 * 6) / 4, dim3(32, 4), 0, stream>>>(yg, opw, x, out);
}
